// NodeEConvModel_80831284510903
// MI455X (gfx1250) — matrix operand order repaired and checked by reading, not compiled, not run on hardware
//
#include <hip/hip_runtime.h>
#include <hip/hip_bf16.h>
#include <math.h>

typedef __bf16 bf16;
typedef bf16 v16bf __attribute__((ext_vector_type(16)));
typedef bf16 v8bf  __attribute__((ext_vector_type(8)));
typedef bf16 v2bf  __attribute__((ext_vector_type(2)));
typedef float v8f  __attribute__((ext_vector_type(8)));

// ---------- helpers ----------

__device__ inline bf16 f2bf(float f) { return (bf16)f; }   // native v_cvt

__device__ inline float lrelu(float v, float s) { return v >= 0.0f ? v : s * v; }

// float atomic max via monotone int/uint ordering trick
__device__ inline void atomicMaxF(float* addr, float val) {
    if (val >= 0.0f) {
        atomicMax((int*)addr, __float_as_int(val));
    } else {
        atomicMin((unsigned int*)addr, __float_as_uint(val));
    }
}

// ---------- WMMA fragment loaders (contiguous k = 16*half + e order, wave32) ----------
// A tile in LDS is row-major [16][KP] bf16. A lane's 16 elements are
// one contiguous 16-element run: K in [kb*32+grp*16, +16) (matches B fragment order).
template<int KP>
__device__ inline v16bf load_a_frag(const bf16* tA, int kb, int lane) {
    int grp = (lane >> 4) & 1;
    int m = lane & 15;
    const bf16* row = tA + m * KP + kb * 32 + grp * 16;
    v8bf lo = *(const v8bf*)row;          // 16B aligned
    v8bf hi = *(const v8bf*)(row + 8);    // 16B aligned
    return __builtin_shufflevector(lo, hi, 0,1,2,3,4,5,6,7,8,9,10,11,12,13,14,15);
}

// B weights pre-swizzled fragment-major: Wf[(kb*(NO/16)+nb)*512 + lane*16 + e]
// = W[kb*32 + (lane>>4)*16 + e][nb*16 + (lane&15)]. One 32B contiguous load per lane.
template<int NO>
__device__ inline v16bf load_b_frag(const bf16* Wf, int kb, int nb, int lane) {
    const bf16* p = Wf + (((size_t)kb * (NO/16) + nb) << 9) + (lane << 4);
    return *(const v16bf*)p;              // 32B aligned -> 2x global_load_b128
}

// D = A(16xKP) * W(KPxNO), f32 accum. acc[nb] is a 16x16 C/D fragment.
template<int KP, int NO>
__device__ inline void wmma_gemm(const bf16* tA, const bf16* Wf, v8f* acc, int lane) {
#pragma unroll
    for (int nb = 0; nb < NO/16; ++nb)
#pragma unroll
        for (int i = 0; i < 8; ++i) acc[nb][i] = 0.0f;
#pragma unroll
    for (int kb = 0; kb < KP/32; ++kb) {
        v16bf a = load_a_frag<KP>(tA, kb, lane);
#pragma unroll
        for (int nb = 0; nb < NO/16; ++nb) {
            v16bf b = load_b_frag<NO>(Wf, kb, nb, lane);
            acc[nb] = __builtin_amdgcn_wmma_f32_16x16x32_bf16(
                false, a, false, b, (short)0, acc[nb], false, false);
        }
    }
}

// C fragment (M = r + 8*grp, N = lane&15) -> bias + leaky-relu -> bf16 LDS tile
template<int NO, int PITCH>
__device__ inline void store_act_lds(const v8f* acc, bf16* tH, const float* bias,
                                     float slope, int lane) {
    int grp = (lane >> 4) & 1;
    int n = lane & 15;
#pragma unroll
    for (int nb = 0; nb < NO/16; ++nb) {
        float bv = bias[nb*16 + n];
#pragma unroll
        for (int r = 0; r < 8; ++r) {
            float v = lrelu(acc[nb][r] + bv, slope);
            tH[(r + 8*grp) * PITCH + nb*16 + n] = f2bf(v);
        }
    }
}

// ---------- small utility kernels ----------

__global__ void fill_kernel(float* p, float v, int n) {
    int i = blockIdx.x * blockDim.x + threadIdx.x;
    if (i < n) p[i] = v;
}

__global__ void count_kernel(const int* __restrict__ src, const int* __restrict__ dst,
                             float* cnt_src, float* cnt_dst, int E) {
    int e = blockIdx.x * blockDim.x + threadIdx.x;
    if (e >= E) return;
    atomicAdd(cnt_src + src[e], 1.0f);
    atomicAdd(cnt_dst + dst[e], 1.0f);
}

// f32 [kin][nout] -> bf16 fragment-major swizzled, zero padded to [kpad][nout]
__global__ void convert_swizzle(const float* __restrict__ s, bf16* __restrict__ d,
                                int kin, int kpad, int nout) {
    int i = blockIdx.x * blockDim.x + threadIdx.x;
    if (i >= kpad * nout) return;
    int frag = i >> 9;            // 512 elements per (kb,nb) fragment
    int rem  = i & 511;
    int lane = rem >> 4;
    int e    = rem & 15;
    int nbc  = nout >> 4;
    int kb   = frag / nbc;
    int nb   = frag - kb * nbc;
    int k = kb * 32 + ((lane >> 4) << 4) + e;
    int n = nb * 16 + (lane & 15);
    d[i] = (k < kin) ? f2bf(s[k * nout + n]) : f2bf(0.0f);
}

__global__ void finalize_kernel(const float* __restrict__ agg, const float* __restrict__ cnt,
                                float* __restrict__ xout, int n, int h2) {
    int i = blockIdx.x * blockDim.x + threadIdx.x;
    if (i >= n * h2) return;
    xout[i] = (cnt[i / h2] > 0.0f) ? agg[i] : 0.0f;
}

// ---------- EdgeConv: msg = mlp(cat[x_i, x_j - x_i]); max-agg at dst ----------
// DIN in-features, KP = padded 2*DIN, H1 hidden, H2 out. 4 waves/block, 1 tile (16 edges)/wave.
template<int DIN, int KP, int H1, int H2>
__global__ __launch_bounds__(128) void edgeconv_kernel(
    const float* __restrict__ x, const int* __restrict__ src, const int* __restrict__ dst,
    const bf16* __restrict__ Wa, const float* __restrict__ ba,
    const bf16* __restrict__ Wb, const float* __restrict__ bb,
    float* __restrict__ agg, int nTiles, int E)
{
    __shared__ bf16 sA[4][16 * KP];
    __shared__ bf16 sH[4][16 * H1];
    int wave = threadIdx.x >> 5;
    int lane = threadIdx.x & 31;
    int tile = blockIdx.x * 4 + wave;
    bool valid = tile < nTiles;
    if (!valid) tile = nTiles - 1;
    int e0 = tile * 16;
    bf16* tA = sA[wave];
    bf16* tH = sH[wave];

    {   // gather: lane handles edge (lane&15); half 0 writes x_i cols, half 1 writes (x_j-x_i)+pad
        int m = lane & 15;
        int half = lane >> 4;
        int e = e0 + m; if (e >= E) e = E - 1;
        int i = dst[e];
        bf16* row = tA + m * KP;
        const float4* xi4 = (const float4*)(x + (size_t)i * DIN);
        if (half == 0) {
#pragma unroll
            for (int c4 = 0; c4 < DIN/4; ++c4) {
                float4 a = xi4[c4];
                v2bf p0 = { f2bf(a.x), f2bf(a.y) };
                v2bf p1 = { f2bf(a.z), f2bf(a.w) };
                *(v2bf*)(row + 4*c4)     = p0;
                *(v2bf*)(row + 4*c4 + 2) = p1;
            }
        } else {
            int j = src[e];
            const float4* xj4 = (const float4*)(x + (size_t)j * DIN);
#pragma unroll
            for (int c4 = 0; c4 < DIN/4; ++c4) {
                float4 a = xi4[c4], b = xj4[c4];
                v2bf p0 = { f2bf(b.x - a.x), f2bf(b.y - a.y) };
                v2bf p1 = { f2bf(b.z - a.z), f2bf(b.w - a.w) };
                *(v2bf*)(row + DIN + 4*c4)     = p0;
                *(v2bf*)(row + DIN + 4*c4 + 2) = p1;
            }
#pragma unroll
            for (int c = 2*DIN; c < KP; c += 2) {
                v2bf z = { f2bf(0.0f), f2bf(0.0f) };
                *(v2bf*)(row + c) = z;
            }
        }
    }
    __syncthreads();

    v8f acc1[H1 / 16];
    wmma_gemm<KP, H1>(tA, Wa, acc1, lane);
    store_act_lds<H1, H1>(acc1, tH, ba, 0.1f, lane);
    __syncthreads();

    v8f acc2[H2 / 16];
    wmma_gemm<H1, H2>(tH, Wb, acc2, lane);

    int grp = (lane >> 4) & 1;
    int n = lane & 15;
#pragma unroll
    for (int nb = 0; nb < H2 / 16; ++nb) {
        float bv = bb[nb*16 + n];
#pragma unroll
        for (int r = 0; r < 8; ++r) {
            int m = r + 8 * grp;
            int e = e0 + m;
            if (valid && e < E) {
                float v = lrelu(acc2[nb][r] + bv, 0.1f);
                atomicMaxF(agg + (size_t)dst[e] * H2 + nb*16 + n, v);
            }
        }
    }
}

// ---------- MetaLayer edge + node models fused ----------
__global__ __launch_bounds__(128) void meta_kernel(
    const float* __restrict__ x3, const int* __restrict__ src, const int* __restrict__ dst,
    const bf16* __restrict__ We1, const float* __restrict__ be1,
    const bf16* __restrict__ We2, const float* __restrict__ be2,
    const bf16* __restrict__ Wn1, const float* __restrict__ bn1,
    const bf16* __restrict__ Wn2, const float* __restrict__ bn2,
    float* __restrict__ sum, int nTiles, int E)
{
    __shared__ bf16 sA[4][16 * 128];   // cat([x[row], x[col]])
    __shared__ bf16 sH[4][16 * 64];    // hidden
    __shared__ bf16 sB[4][16 * 96];    // cat([x[col], e]) padded 80->96
    int wave = threadIdx.x >> 5;
    int lane = threadIdx.x & 31;
    int tile = blockIdx.x * 4 + wave;
    bool valid = tile < nTiles;
    if (!valid) tile = nTiles - 1;
    int e0 = tile * 16;
    bf16* tA = sA[wave];
    bf16* tH = sH[wave];
    bf16* tB = sB[wave];

    {   // gather: half 0 writes x[row] -> tA[0:64); half 1 writes x[col] -> tA[64:128), tB[0:64), pads tB
        int m = lane & 15;
        int half = lane >> 4;
        int e = e0 + m; if (e >= E) e = E - 1;
        bf16* rowA = tA + m * 128;
        bf16* rowB = tB + m * 96;
        if (half == 0) {
            int r = src[e];
            const float4* xr4 = (const float4*)(x3 + (size_t)r * 64);
#pragma unroll
            for (int c4 = 0; c4 < 16; ++c4) {
                float4 a = xr4[c4];
                v2bf p0 = { f2bf(a.x), f2bf(a.y) };
                v2bf p1 = { f2bf(a.z), f2bf(a.w) };
                *(v2bf*)(rowA + 4*c4)     = p0;
                *(v2bf*)(rowA + 4*c4 + 2) = p1;
            }
        } else {
            int c = dst[e];
            const float4* xc4 = (const float4*)(x3 + (size_t)c * 64);
#pragma unroll
            for (int c4 = 0; c4 < 16; ++c4) {
                float4 a = xc4[c4];
                v2bf p0 = { f2bf(a.x), f2bf(a.y) };
                v2bf p1 = { f2bf(a.z), f2bf(a.w) };
                *(v2bf*)(rowA + 64 + 4*c4)     = p0;
                *(v2bf*)(rowA + 64 + 4*c4 + 2) = p1;
                *(v2bf*)(rowB + 4*c4)          = p0;
                *(v2bf*)(rowB + 4*c4 + 2)      = p1;
            }
#pragma unroll
            for (int k = 80; k < 96; k += 2) {
                v2bf z = { f2bf(0.0f), f2bf(0.0f) };
                *(v2bf*)(rowB + k) = z;
            }
        }
    }
    __syncthreads();

    // edge model: lrelu(cat @ We1 + be1, .12) @ We2 + be2   -> e[16]
    v8f accH[4];
    wmma_gemm<128, 64>(tA, We1, accH, lane);
    store_act_lds<64, 64>(accH, tH, be1, 0.12f, lane);
    __syncthreads();

    v8f accE[1];
    wmma_gemm<64, 16>(tH, We2, accE, lane);
    {
        int grp = (lane >> 4) & 1;
        int n = lane & 15;
        float bv = be2[n];
#pragma unroll
        for (int r = 0; r < 8; ++r)
            tB[(r + 8*grp) * 96 + 64 + n] = f2bf(accE[0][r] + bv);  // no activation
    }
    __syncthreads();

    // node model: lrelu(cat([x[col], e]) @ Wn1 + bn1, .12) @ Wn2 + bn2 -> scatter-add at src
    v8f accN1[4];
    wmma_gemm<96, 64>(tB, Wn1, accN1, lane);
    store_act_lds<64, 64>(accN1, tH, bn1, 0.12f, lane);
    __syncthreads();

    v8f accN2[2];
    wmma_gemm<64, 32>(tH, Wn2, accN2, lane);

    int grp = (lane >> 4) & 1;
    int n = lane & 15;
#pragma unroll
    for (int nb = 0; nb < 2; ++nb) {
        float bv = bn2[nb*16 + n];
#pragma unroll
        for (int r = 0; r < 8; ++r) {
            int m = r + 8 * grp;
            int e = e0 + m;
            if (valid && e < E) {
                float v = accN2[nb][r] + bv;
                atomicAdd(sum + (size_t)src[e] * 32 + nb*16 + n, v);
            }
        }
    }
}

// ---------- per-node head: mean -> 32->16->2 -> log_softmax ----------
__global__ void node_out_kernel(const float* __restrict__ sum, const float* __restrict__ cnt_src,
                                const float* __restrict__ wm1, const float* __restrict__ bm1,
                                const float* __restrict__ wm2, const float* __restrict__ bm2,
                                float* __restrict__ out, int n)
{
    int i = blockIdx.x * blockDim.x + threadIdx.x;
    if (i >= n) return;
    float c = fmaxf(cnt_src[i], 1.0f);
    float inv = 1.0f / c;
    float mean[32];
#pragma unroll
    for (int k = 0; k < 32; ++k) mean[k] = sum[(size_t)i * 32 + k] * inv;
    float h[16];
#pragma unroll
    for (int o = 0; o < 16; ++o) {
        float acc = bm1[o];
#pragma unroll
        for (int k = 0; k < 32; ++k) acc += mean[k] * wm1[k*16 + o];
        h[o] = lrelu(acc, 0.12f);
    }
    float o0 = bm2[0], o1 = bm2[1];
#pragma unroll
    for (int k = 0; k < 16; ++k) { o0 += h[k] * wm2[k*2 + 0]; o1 += h[k] * wm2[k*2 + 1]; }
    float mx = fmaxf(o0, o1);
    float lse = mx + logf(expf(o0 - mx) + expf(o1 - mx));
    out[(size_t)i*2 + 0] = o0 - lse;
    out[(size_t)i*2 + 1] = o1 - lse;
}

// ---------- host ----------
extern "C" void kernel_launch(void* const* d_in, const int* in_sizes, int n_in,
                              void* d_out, int out_size, void* d_ws, size_t ws_size,
                              hipStream_t stream) {
    (void)n_in; (void)out_size; (void)ws_size;
    const int N = in_sizes[0] / 4;
    const int E = in_sizes[1] / 2;
    const float* x0  = (const float*)d_in[0];
    const int*   src = (const int*)d_in[1];
    const int*   dst = src + E;
    const float* w1a=(const float*)d_in[2];  const float* b1a=(const float*)d_in[3];
    const float* w1b=(const float*)d_in[4];  const float* b1b=(const float*)d_in[5];
    const float* w2a=(const float*)d_in[6];  const float* b2a=(const float*)d_in[7];
    const float* w2b=(const float*)d_in[8];  const float* b2b=(const float*)d_in[9];
    const float* w3a=(const float*)d_in[10]; const float* b3a=(const float*)d_in[11];
    const float* w3b=(const float*)d_in[12]; const float* b3b=(const float*)d_in[13];
    const float* we1=(const float*)d_in[14]; const float* be1=(const float*)d_in[15];
    const float* we2=(const float*)d_in[16]; const float* be2=(const float*)d_in[17];
    const float* wn1=(const float*)d_in[18]; const float* bn1=(const float*)d_in[19];
    const float* wn2=(const float*)d_in[20]; const float* bn2=(const float*)d_in[21];
    const float* wm1=(const float*)d_in[22]; const float* bm1=(const float*)d_in[23];
    const float* wm2=(const float*)d_in[24]; const float* bm2=(const float*)d_in[25];

    char* ws = (char*)d_ws;
    size_t off = 0;
    auto alloc = [&](size_t bytes) -> char* {
        char* p = ws + off;
        off = (off + bytes + 255) & ~(size_t)255;
        return p;
    };
    bf16* W1a = (bf16*)alloc(32*32*2);   bf16* W1b = (bf16*)alloc(32*16*2);
    bf16* W2a = (bf16*)alloc(32*64*2);   bf16* W2b = (bf16*)alloc(64*32*2);
    bf16* W3a = (bf16*)alloc(64*128*2);  bf16* W3b = (bf16*)alloc((size_t)128*64*2);
    bf16* We1 = (bf16*)alloc((size_t)128*64*2); bf16* We2 = (bf16*)alloc(64*16*2);
    bf16* Wn1 = (bf16*)alloc(96*64*2);   bf16* Wn2 = (bf16*)alloc(64*32*2);
    float* cnt_dst = (float*)alloc((size_t)N*4);
    float* cnt_src = (float*)alloc((size_t)N*4);
    float* x1  = (float*)alloc((size_t)N*16*4);
    float* x2  = (float*)alloc((size_t)N*32*4);
    float* x3  = (float*)alloc((size_t)N*64*4);
    float* agg = (float*)alloc((size_t)N*64*4);
    float* sum = (float*)alloc((size_t)N*32*4);

    auto cdiv = [](int a, int b) { return (a + b - 1) / b; };

    convert_swizzle<<<cdiv(32*32,256),256,0,stream>>>(w1a, W1a, 8, 32, 32);
    convert_swizzle<<<cdiv(32*16,256),256,0,stream>>>(w1b, W1b, 32, 32, 16);
    convert_swizzle<<<cdiv(32*64,256),256,0,stream>>>(w2a, W2a, 32, 32, 64);
    convert_swizzle<<<cdiv(64*32,256),256,0,stream>>>(w2b, W2b, 64, 64, 32);
    convert_swizzle<<<cdiv(64*128,256),256,0,stream>>>(w3a, W3a, 64, 64, 128);
    convert_swizzle<<<cdiv(128*64,256),256,0,stream>>>(w3b, W3b, 128, 128, 64);
    convert_swizzle<<<cdiv(128*64,256),256,0,stream>>>(we1, We1, 128, 128, 64);
    convert_swizzle<<<cdiv(64*16,256),256,0,stream>>>(we2, We2, 64, 64, 16);
    convert_swizzle<<<cdiv(96*64,256),256,0,stream>>>(wn1, Wn1, 80, 96, 64);
    convert_swizzle<<<cdiv(64*32,256),256,0,stream>>>(wn2, Wn2, 64, 64, 32);

    fill_kernel<<<cdiv(N,256),256,0,stream>>>(cnt_dst, 0.0f, N);
    fill_kernel<<<cdiv(N,256),256,0,stream>>>(cnt_src, 0.0f, N);
    count_kernel<<<cdiv(E,256),256,0,stream>>>(src, dst, cnt_src, cnt_dst, E);

    const int nTiles  = cdiv(E, 16);
    const int nBlocks = cdiv(nTiles, 4);
    const float NEG = -3.0e38f;

    // EdgeConv 1: 4 -> (8->32->16)
    fill_kernel<<<cdiv(N*16,256),256,0,stream>>>(agg, NEG, N*16);
    edgeconv_kernel<4,32,32,16><<<nBlocks,128,0,stream>>>(x0, src, dst, W1a, b1a, W1b, b1b, agg, nTiles, E);
    finalize_kernel<<<cdiv(N*16,256),256,0,stream>>>(agg, cnt_dst, x1, N, 16);

    // EdgeConv 2: 16 -> (32->64->32)
    fill_kernel<<<cdiv(N*32,256),256,0,stream>>>(agg, NEG, N*32);
    edgeconv_kernel<16,32,64,32><<<nBlocks,128,0,stream>>>(x1, src, dst, W2a, b2a, W2b, b2b, agg, nTiles, E);
    finalize_kernel<<<cdiv(N*32,256),256,0,stream>>>(agg, cnt_dst, x2, N, 32);

    // EdgeConv 3: 32 -> (64->128->64)
    fill_kernel<<<cdiv(N*64,256),256,0,stream>>>(agg, NEG, N*64);
    edgeconv_kernel<32,64,128,64><<<nBlocks,128,0,stream>>>(x2, src, dst, W3a, b3a, W3b, b3b, agg, nTiles, E);
    finalize_kernel<<<cdiv(N*64,256),256,0,stream>>>(agg, cnt_dst, x3, N, 64);

    // MetaLayer edge + node models
    fill_kernel<<<cdiv(N*32,256),256,0,stream>>>(sum, 0.0f, N*32);
    meta_kernel<<<nBlocks,128,0,stream>>>(x3, src, dst, We1, be1, We2, be2,
                                          Wn1, bn1, Wn2, bn2, sum, nTiles, E);

    node_out_kernel<<<cdiv(N,256),256,0,stream>>>(sum, cnt_src, wm1, bm1, wm2, bm2,
                                                  (float*)d_out, N);
}
